// DynamicResidualStageWrapper_38362647888581
// MI455X (gfx1250) — matrix operand order repaired and checked by reading, not compiled, not run on hardware
//
#include <hip/hip_runtime.h>
#include <hip/hip_bf16.h>
#include <math.h>

#define NUMBLK 18
#define CDIM 512
#define HID 128
#define BATCH 32
#define TOK 196                    // 14*14
#define M_TOTAL (BATCH * TOK)      // 6272
#define NANCH 3

typedef __attribute__((ext_vector_type(16))) __bf16 v16bf;
typedef __attribute__((ext_vector_type(8)))  __bf16 v8bf;
typedef __attribute__((ext_vector_type(8)))  float  v8f;

__device__ __forceinline__ float gelu_tanh(float x) {
    const float k0 = 0.7978845608028654f;   // sqrt(2/pi)
    const float k1 = 0.044715f;
    return 0.5f * x * (1.0f + tanhf(k0 * (x + k1 * x * x * x)));
}

// ---------------------------------------------------------------------------
// GEMM + bias + GELU:  Y[M,512] = gelu(X[M,512] @ W[512,512] + b)
// 128x128 tile / block (256 thr = 8 waves), K-step 32, bf16 hi/lo split (3 WMMA)
// ---------------------------------------------------------------------------
#define TM 128
#define TN 128
#define TK 32
#define ASTR 40   // padded LDS row stride in elements (80B: 16B aligned, conflict-spread)

__global__ __launch_bounds__(256) void gemm_gelu_kernel(
    const float* __restrict__ X, const float* __restrict__ W,
    const float* __restrict__ bias, float* __restrict__ Y)
{
    __shared__ __align__(16) __bf16 sAh[TM * ASTR];
    __shared__ __align__(16) __bf16 sAl[TM * ASTR];
    __shared__ __align__(16) __bf16 sBh[TN * ASTR];   // transposed: [n][k]
    __shared__ __align__(16) __bf16 sBl[TN * ASTR];

    const int tid  = threadIdx.x;
    const int m0   = blockIdx.x * TM;
    const int n0   = blockIdx.y * TN;

    const int wid  = tid >> 5;
    const int lane = tid & 31;
    const int wm   = wid >> 2;     // 0..1  -> 64 rows
    const int wn   = wid & 3;      // 0..3  -> 32 cols
    const int half = lane >> 4;    // lane group
    const int lr   = lane & 15;

    v8f acc[4][2];
#pragma unroll
    for (int mt = 0; mt < 4; ++mt)
#pragma unroll
        for (int nt = 0; nt < 2; ++nt)
#pragma unroll
            for (int v = 0; v < 8; ++v) acc[mt][nt][v] = 0.0f;

    for (int kk = 0; kk < CDIM; kk += TK) {
        __syncthreads();
        // ---- stage A tile: X[m0..m0+128, kk..kk+32], split fp32 -> bf16 hi/lo
#pragma unroll
        for (int j = 0; j < 4; ++j) {
            int idx4 = j * 256 + tid;          // 1024 float4 total
            int row  = idx4 >> 3;              // 8 float4 per row of 32
            int c4   = (idx4 & 7) * 4;
            const float4 xv = *(const float4*)(X + (size_t)(m0 + row) * CDIM + kk + c4);
            float v[4] = {xv.x, xv.y, xv.z, xv.w};
#pragma unroll
            for (int u = 0; u < 4; ++u) {
                __bf16 h = (__bf16)v[u];
                __bf16 l = (__bf16)(v[u] - (float)h);
                sAh[row * ASTR + c4 + u] = h;
                sAl[row * ASTR + c4 + u] = l;
            }
        }
        // ---- stage B tile: W[kk..kk+32, n0..n0+128], transpose into [n][k]
#pragma unroll
        for (int j = 0; j < 4; ++j) {
            int idx4 = j * 256 + tid;
            int krow = idx4 >> 5;              // 32 float4 per k-row of 128
            int c4   = (idx4 & 31) * 4;
            const float4 wv = *(const float4*)(W + (size_t)(kk + krow) * CDIM + n0 + c4);
            float v[4] = {wv.x, wv.y, wv.z, wv.w};
#pragma unroll
            for (int u = 0; u < 4; ++u) {
                __bf16 h = (__bf16)v[u];
                __bf16 l = (__bf16)(v[u] - (float)h);
                sBh[(c4 + u) * ASTR + krow] = h;
                sBl[(c4 + u) * ASTR + krow] = l;
            }
        }
        __syncthreads();

        // ---- B fragments: lane holds col n=lr; e0..7 -> K in [8h,8h+8), e8..15 -> [16+8h,..)
        v16bf Bh[2], Bl[2];
#pragma unroll
        for (int nt = 0; nt < 2; ++nt) {
            int nb = (wn * 32 + nt * 16 + lr) * ASTR + 8 * half;
            v8bf b0 = *reinterpret_cast<const v8bf*>(sBh + nb);
            v8bf b1 = *reinterpret_cast<const v8bf*>(sBh + nb + 16);
            Bh[nt] = __builtin_shufflevector(b0, b1, 0,1,2,3,4,5,6,7,8,9,10,11,12,13,14,15);
            v8bf c0 = *reinterpret_cast<const v8bf*>(sBl + nb);
            v8bf c1 = *reinterpret_cast<const v8bf*>(sBl + nb + 16);
            Bl[nt] = __builtin_shufflevector(c0, c1, 0,1,2,3,4,5,6,7,8,9,10,11,12,13,14,15);
        }
        // ---- A fragments: lane holds row m=lr; e0..7 -> K in [8h,8h+8), e8..15 -> [16+8h,..)
#pragma unroll
        for (int mt = 0; mt < 4; ++mt) {
            int ab = (wm * 64 + mt * 16 + lr) * ASTR + 8 * half;
            v8bf a0 = *reinterpret_cast<const v8bf*>(sAh + ab);
            v8bf a1 = *reinterpret_cast<const v8bf*>(sAh + ab + 16);
            v16bf Ah = __builtin_shufflevector(a0, a1, 0,1,2,3,4,5,6,7,8,9,10,11,12,13,14,15);
            v8bf a2 = *reinterpret_cast<const v8bf*>(sAl + ab);
            v8bf a3 = *reinterpret_cast<const v8bf*>(sAl + ab + 16);
            v16bf Al = __builtin_shufflevector(a2, a3, 0,1,2,3,4,5,6,7,8,9,10,11,12,13,14,15);
#pragma unroll
            for (int nt = 0; nt < 2; ++nt) {
                acc[mt][nt] = __builtin_amdgcn_wmma_f32_16x16x32_bf16(
                    false, Ah, false, Bh[nt], (short)0, acc[mt][nt], false, false);
                acc[mt][nt] = __builtin_amdgcn_wmma_f32_16x16x32_bf16(
                    false, Ah, false, Bl[nt], (short)0, acc[mt][nt], false, false);
                acc[mt][nt] = __builtin_amdgcn_wmma_f32_16x16x32_bf16(
                    false, Al, false, Bh[nt], (short)0, acc[mt][nt], false, false);
            }
        }
    }

    // ---- epilogue: bias + GELU, C/D layout: VGPR v -> row v + 8*half, lane lr -> col
#pragma unroll
    for (int mt = 0; mt < 4; ++mt)
#pragma unroll
        for (int nt = 0; nt < 2; ++nt) {
            int n = n0 + wn * 32 + nt * 16 + lr;
            float bn = bias[n];
#pragma unroll
            for (int v = 0; v < 8; ++v) {
                int m = m0 + wm * 64 + mt * 16 + v + 8 * half;
                Y[(size_t)m * CDIM + n] = gelu_tanh(acc[mt][nt][v] + bn);
            }
        }
}

// ---------------------------------------------------------------------------
// pooled[b][c] = mean over 196 tokens of nx
// ---------------------------------------------------------------------------
__global__ __launch_bounds__(256) void pool_kernel(
    const float* __restrict__ nx, float* __restrict__ pooled)
{
    int b = blockIdx.x;
    for (int c = threadIdx.x; c < CDIM; c += 256) {
        const float* p = nx + (size_t)b * TOK * CDIM + c;
        float s = 0.0f;
        for (int r = 0; r < TOK; ++r) s += p[(size_t)r * CDIM];
        pooled[b * CDIM + c] = s * (1.0f / (float)TOK);
    }
}

// ---------------------------------------------------------------------------
// gating MLP + softmax over anchors; one block per image
// ---------------------------------------------------------------------------
__global__ __launch_bounds__(256) void gate_kernel(
    const float* __restrict__ pooled,
    const float* __restrict__ fc1w, const float* __restrict__ fc1b,
    const float* __restrict__ fc2w, const float* __restrict__ fc2b,
    float* __restrict__ gates)
{
    __shared__ float ps[CDIM];
    __shared__ float hs[HID];
    __shared__ float ls[NANCH * CDIM];
    int b = blockIdx.x, t = threadIdx.x;
    ps[t]       = pooled[b * CDIM + t];
    ps[t + 256] = pooled[b * CDIM + t + 256];
    __syncthreads();
    if (t < HID) {
        float s = fc1b[t];
        for (int k = 0; k < CDIM; ++k) s += ps[k] * fc1w[k * HID + t];
        hs[t] = gelu_tanh(s);
    }
    __syncthreads();
    for (int o = t; o < NANCH * CDIM; o += 256) {
        float s = fc2b[o];
        for (int k = 0; k < HID; ++k) s += hs[k] * fc2w[k * (NANCH * CDIM) + o];
        ls[o] = s;
    }
    __syncthreads();
    for (int c = t; c < CDIM; c += 256) {
        float l0 = ls[c], l1 = ls[CDIM + c], l2 = ls[2 * CDIM + c];
        float mx = fmaxf(l0, fmaxf(l1, l2));
        float e0 = expf(l0 - mx), e1 = expf(l1 - mx), e2 = expf(l2 - mx);
        float inv = 1.0f / (e0 + e1 + e2);
        gates[((size_t)b * NANCH + 0) * CDIM + c] = e0 * inv;
        gates[((size_t)b * NANCH + 1) * CDIM + c] = e1 * inv;
        gates[((size_t)b * NANCH + 2) * CDIM + c] = e2 * inv;
    }
}

// ---------------------------------------------------------------------------
// out = nx + gamma * sum_a gates[b,a,c] * anchor_a   (float4 per thread)
// ---------------------------------------------------------------------------
__global__ __launch_bounds__(256) void route_kernel(
    const float* nx,
    const float* __restrict__ a0, const float* __restrict__ a1,
    const float* __restrict__ a2, const float* __restrict__ gates,
    const float* __restrict__ gma, float* outp)
{
    int g = blockIdx.x * 256 + threadIdx.x;
    int m = g >> 7;                 // 128 float4 per row of 512
    int c = (g & 127) * 4;
    int b = m / TOK;
    float gamma = *gma;
    size_t off = (size_t)m * CDIM + c;
    float4 vx = *(const float4*)(nx + off);
    float4 v0 = *(const float4*)(a0 + off);
    float4 v1 = *(const float4*)(a1 + off);
    float4 v2 = *(const float4*)(a2 + off);
    size_t gb = (size_t)b * NANCH * CDIM + c;
    float4 g0 = *(const float4*)(gates + gb);
    float4 g1 = *(const float4*)(gates + gb + CDIM);
    float4 g2 = *(const float4*)(gates + gb + 2 * CDIM);
    float4 r;
    r.x = vx.x + gamma * (g0.x * v0.x + g1.x * v1.x + g2.x * v2.x);
    r.y = vx.y + gamma * (g0.y * v0.y + g1.y * v1.y + g2.y * v2.y);
    r.z = vx.z + gamma * (g0.z * v0.z + g1.z * v1.z + g2.z * v2.z);
    r.w = vx.w + gamma * (g0.w * v0.w + g1.w * v1.w + g2.w * v2.w);
    *(float4*)(outp + off) = r;
}

// ---------------------------------------------------------------------------
extern "C" void kernel_launch(void* const* d_in, const int* in_sizes, int n_in,
                              void* d_out, int out_size, void* d_ws, size_t ws_size,
                              hipStream_t stream)
{
    (void)in_sizes; (void)n_in; (void)out_size; (void)ws_size;
    const float* x       = (const float*)d_in[0];
    const float* block_w = (const float*)d_in[1];   // [18][512][512]
    const float* block_b = (const float*)d_in[2];   // [18][512]
    const float* fc1_w   = (const float*)d_in[3];   // [3][512][128]
    const float* fc1_b   = (const float*)d_in[4];   // [3][128]
    const float* fc2_w   = (const float*)d_in[5];   // [3][128][1536]
    const float* fc2_b   = (const float*)d_in[6];   // [3][1536]
    const float* gammas  = (const float*)d_in[7];   // [3]
    float* out = (float*)d_out;

    const size_t NT = (size_t)M_TOTAL * CDIM;       // 3,211,264 floats
    float* ws     = (float*)d_ws;
    float* bufs[2] = { ws, ws + NT };               // ping / pong
    float* anc[3]  = { ws + 2 * NT, ws + 3 * NT, ws + 4 * NT };
    float* pooled  = ws + 5 * NT;                   // [32][512]
    float* gates   = pooled + (size_t)BATCH * CDIM; // [32][3][512]

    dim3 ggrid(M_TOTAL / TM, CDIM / TN);            // 49 x 4
    const int rgrid = (M_TOTAL * CDIM / 4) / 256;   // 3136

    const float* cur = x;
    int pp = 0;
    for (int i = 0; i < NUMBLK; ++i) {
        float* dst;
        if      (i == 1) dst = anc[0];
        else if (i == 4) dst = anc[1];
        else if (i == 9) dst = anc[2];
        else { dst = bufs[pp]; pp ^= 1; }

        gemm_gelu_kernel<<<ggrid, 256, 0, stream>>>(
            cur, block_w + (size_t)i * CDIM * CDIM, block_b + (size_t)i * CDIM, dst);

        int t = (i == 11) ? 0 : (i == 14) ? 1 : (i == 17) ? 2 : -1;
        if (t >= 0) {
            pool_kernel<<<BATCH, 256, 0, stream>>>(dst, pooled);
            gate_kernel<<<BATCH, 256, 0, stream>>>(pooled,
                fc1_w + (size_t)t * CDIM * HID, fc1_b + (size_t)t * HID,
                fc2_w + (size_t)t * HID * NANCH * CDIM, fc2_b + (size_t)t * NANCH * CDIM,
                gates);
            float* rout = (i == 17) ? out : dst;    // last block writes final output
            route_kernel<<<rgrid, 256, 0, stream>>>(
                dst, anc[0], anc[1], anc[2], gates, gammas + t, rout);
        }
        cur = dst;
    }
}
